// SE3Transformer_18502719111870
// MI455X (gfx1250) — compile-verified, matrix operand order checked by reading, not run on hardware
//
#include <hip/hip_runtime.h>
#include <math.h>

// ---------------- problem constants ----------------
#define NNODES 2048
#define NDEG   16
#define NEDGES (NNODES*NDEG)
#define NODE_FEAT 32
#define NCH    16
#define MO_MID 4
#define NHEADS 4
#define OUTD   64
#define NLAYERS 2
#define LMAXSH 6
#define PI_D 3.14159265358979323846
#define GMT 4   // M-tiles per GEMM wave (amortizes B-fragment load)

typedef __attribute__((ext_vector_type(16))) _Float16 v16h;
typedef __attribute__((ext_vector_type(8)))  float    v8f;
typedef __attribute__((ext_vector_type(4)))  unsigned int u32x4;
typedef __attribute__((ext_vector_type(8)))  int      i32x8;
typedef __attribute__((ext_vector_type(4)))  int      i32x4;

__constant__ double c_fact[14] = {1.0,1.0,2.0,6.0,24.0,120.0,720.0,5040.0,40320.0,
                                  362880.0,3628800.0,39916800.0,479001600.0,6227020800.0};

// offset (in floats) of QJ blocks for pair (di,dq); J blocks stored ascending.
__host__ __device__ inline int qj_pair_off(int di, int dq) {
  int off = 0;
  for (int a = 0; a < 4; a++)
    for (int b = 0; b < 4; b++) {
      if (a == di && b == dq) return off;
      int lo = a > b ? a - b : b - a;
      for (int J = lo; J <= a + b; J++) off += (2*a+1)*(2*J+1)*(2*b+1);
    }
  return off;
}

// ---------------- QJ (real Clebsch-Gordan) on device ----------------
struct DC { double re, im; };
__device__ inline DC cmul(DC a, DC b) { DC r; r.re = a.re*b.re - a.im*b.im; r.im = a.re*b.im + a.im*b.re; return r; }

__device__ double d_cg(int j1,int m1,int j2,int m2,int j3,int m3) {
  int dj = j1 - j2; if (dj < 0) dj = -dj;
  if (m1 + m2 != m3 || j3 < dj || j3 > j1 + j2) return 0.0;
  double pre = sqrt((2*j3+1) * c_fact[j1+j2-j3] * c_fact[j1-j2+j3] * c_fact[-j1+j2+j3] / c_fact[j1+j2+j3+1]);
  pre *= sqrt(c_fact[j1+m1]*c_fact[j1-m1]*c_fact[j2+m2]*c_fact[j2-m2]*c_fact[j3+m3]*c_fact[j3-m3]);
  int kmin = 0;
  if (j2 - j3 - m1 > kmin) kmin = j2 - j3 - m1;
  if (j1 + m2 - j3 > kmin) kmin = j1 + m2 - j3;
  int kmax = j1 + j2 - j3;
  if (j1 - m1 < kmax) kmax = j1 - m1;
  if (j2 + m2 < kmax) kmax = j2 + m2;
  double s = 0.0;
  for (int k = kmin; k <= kmax; k++) {
    double t = ((k & 1) ? -1.0 : 1.0) /
      (c_fact[k]*c_fact[j1+j2-j3-k]*c_fact[j1-m1-k]*c_fact[j2+m2-k]*c_fact[j3-j2+m1+k]*c_fact[j3-j1-m2+k]);
    s += t;
  }
  return pre * s;
}

__device__ int r2c_row(int l, int r, int* cols, DC* vals) {
  int m = r - l;
  const double s = 0.7071067811865476;
  if (m == 0) { cols[0] = l; vals[0].re = 1.0; vals[0].im = 0.0; return 1; }
  if (m > 0) {
    double sg = (m & 1) ? -1.0 : 1.0;
    cols[0] = l + m; vals[0].re = sg * s; vals[0].im = 0.0;
    cols[1] = l - m; vals[1].re = s;      vals[1].im = 0.0;
    return 2;
  }
  int am = -m;
  double sg = (am & 1) ? -1.0 : 1.0;
  cols[0] = l - am; vals[0].re = 0.0; vals[0].im = s;
  cols[1] = l + am; vals[1].re = 0.0; vals[1].im = -sg * s;
  return 2;
}

__device__ DC t_entry(int l1,int l2,int l3,int i,int j,int k) {
  int c1[2], c2[2], c3[2]; DC v1[2], v2[2], v3[2];
  int n1 = r2c_row(l1, i, c1, v1);
  int n2 = r2c_row(l2, j, c2, v2);
  int n3 = r2c_row(l3, k, c3, v3);
  DC acc; acc.re = 0.0; acc.im = 0.0;
  for (int a = 0; a < n1; a++)
    for (int b = 0; b < n2; b++)
      for (int c = 0; c < n3; c++) {
        int m1 = c1[a] - l1, m2 = c2[b] - l2, m3 = c3[c] - l3;
        if (m1 + m2 != m3) continue;
        double C = d_cg(l1, m1, l2, m2, l3, m3);
        if (C == 0.0) continue;
        DC p = cmul(v1[a], v2[b]);
        DC q; q.re = v3[c].re; q.im = -v3[c].im;
        p = cmul(p, q);
        acc.re += C * p.re; acc.im += C * p.im;
      }
  return acc;
}

__global__ void k_qj(float* qj) {
  int tid = threadIdx.x;
  int ord = 0, off = 0;
  for (int a = 0; a < 4; a++)
    for (int b = 0; b < 4; b++) {
      int lo = a > b ? a - b : b - a;
      for (int J = lo; J <= a + b; J++) {
        int d1 = 2*a+1, d2 = 2*J+1, d3 = 2*b+1;
        if ((ord % (int)blockDim.x) == tid) {
          double mre = 0.0, mim = 0.0;
          for (int i = 0; i < d1; i++)
            for (int j = 0; j < d2; j++)
              for (int k = 0; k < d3; k++) {
                DC t = t_entry(a, J, b, i, j, k);
                mre = fmax(mre, fabs(t.re)); mim = fmax(mim, fabs(t.im));
              }
          int useIm = (mim > mre);
          for (int i = 0; i < d1; i++)
            for (int j = 0; j < d2; j++)
              for (int k = 0; k < d3; k++) {
                DC t = t_entry(a, J, b, i, j, k);
                qj[off + (i*d2 + j)*d3 + k] = (float)(useIm ? t.im : t.re);
              }
        }
        off += d1*d2*d3; ord++;
      }
    }
}

// ---------------- edge geometry, spherical harmonics, padded feat ----------------
__global__ void k_edge(const float* __restrict__ x, const float* __restrict__ wfeat,
                       const int* __restrict__ src, const int* __restrict__ dst,
                       float* __restrict__ Y, _Float16* __restrict__ featP) {
  int e = blockIdx.x * blockDim.x + threadIdx.x;
  if (e >= NEDGES) return;
  int s = src[e], d = dst[e];
  float dx = x[d*3+0]-x[s*3+0], dy = x[d*3+1]-x[s*3+1], dz = x[d*3+2]-x[s*3+2];
  float r = sqrtf(dx*dx + dy*dy + dz*dz + 1e-12f);
  float ux = dx/r, uy = dy/r, uz = dz/r;
  _Float16* fp = featP + (size_t)e*32;
  fp[0] = (_Float16)wfeat[e*4+0]; fp[1] = (_Float16)wfeat[e*4+1];
  fp[2] = (_Float16)wfeat[e*4+2]; fp[3] = (_Float16)wfeat[e*4+3];
  fp[4] = (_Float16)r;
  for (int i = 5; i < 32; i++) fp[i] = (_Float16)0.f;

  float z = uz;
  float st = sqrtf(fminf(fmaxf(1.f - z*z, 1e-12f), 1.f));
  float phi = atan2f(uy, ux);
  float P[LMAXSH+1][LMAXSH+1];
  P[0][0] = 1.f;
  #pragma unroll
  for (int m = 1; m <= LMAXSH; m++) P[m][m] = -(float)(2*m-1) * st * P[m-1][m-1];
  #pragma unroll
  for (int m = 0; m < LMAXSH; m++) P[m+1][m] = (float)(2*m+1) * z * P[m][m];
  #pragma unroll
  for (int m = 0; m <= LMAXSH; m++)
    for (int l = m+2; l <= LMAXSH; l++)
      P[l][m] = ((float)(2*l-1)*z*P[l-1][m] - (float)(l+m-1)*P[l-2][m]) / (float)(l-m);

  float* Ye = Y + (size_t)e*49;
  for (int l = 0; l <= LMAXSH; l++)
    for (int m = -l; m <= l; m++) {
      int am = m < 0 ? -m : m;
      double Nc = sqrt((2*l+1)/(4.0*PI_D) * c_fact[l-am]/c_fact[l+am]);
      float v;
      if (m > 0)      v = (float)(1.4142135623730951*Nc) * cosf((float)m*phi) * P[l][am];
      else if (m==0)  v = (float)Nc * P[l][0];
      else            v = (float)(1.4142135623730951*Nc) * sinf((float)am*phi) * P[l][am];
      Ye[l*l + (l+m)] = v;
    }
}

// ---------------- weight convert: f32 (rows x ldw) -> padded f16 (rows x 32) ----------------
__global__ void k_cvtw(const float* __restrict__ W, int ldw, int kval, int rows,
                       _Float16* __restrict__ out) {
  int i = blockIdx.x * blockDim.x + threadIdx.x;
  if (i >= rows*32) return;
  int r = i >> 5, k = i & 31;
  out[i] = (k < kval) ? (_Float16)W[(size_t)r*ldw + k] : (_Float16)0.f;
}

// ---------------- WMMA fragment helpers (16x16x32 f16 -> f32) ----------------
__device__ inline v16h load_a_frag(const _Float16* A, int m0) {
  int lane = threadIdx.x & 31;
  int row = m0 + (lane & 15);
  int kh = (lane & 16) ? 8 : 0;
  const _Float16* p = A + (size_t)row*32 + kh;
  v16h a;
  #pragma unroll
  for (int t = 0; t < 8; t++) { a[t] = p[t]; a[8+t] = p[16+t]; }
  return a;
}

// A fragment from f32 pre-activation Z with fused batch-norm + ReLU + f16 convert.
__device__ inline v16h load_a_bn(const float* Z, int m0, const float* stats,
                                 const float* g, const float* be) {
  int lane = threadIdx.x & 31;
  int row = m0 + (lane & 15);
  int kh = (lane & 16) ? 8 : 0;
  const float* p = Z + (size_t)row*32;
  v16h a;
  #pragma unroll
  for (int t = 0; t < 8; t++) {
    int k0 = kh + t, k1 = kh + 16 + t;
    float v0 = (p[k0] - stats[k0]) * stats[32 + k0] * g[k0] + be[k0];
    float v1 = (p[k1] - stats[k1]) * stats[32 + k1] * g[k1] + be[k1];
    a[t]   = (_Float16)fmaxf(v0, 0.f);
    a[8+t] = (_Float16)fmaxf(v1, 0.f);
  }
  return a;
}

// GEMM: Z[m, n] = act(A)[m, :32] @ Wh[n, :32]^T + bias[n]
__global__ void k_gemm16(const _Float16* __restrict__ A16, const float* __restrict__ Zin,
                         const float* __restrict__ stats, const float* __restrict__ g,
                         const float* __restrict__ be,
                         const _Float16* __restrict__ Wh, const float* __restrict__ bias,
                         float* __restrict__ Z, int Nout) {
  int nt = blockIdx.y;
  v16h b = load_a_frag(Wh, nt*16);
  int lane = threadIdx.x & 31;
  int n = nt*16 + (lane & 15);
  float bv = bias[n];
  #pragma unroll
  for (int it = 0; it < GMT; it++) {
    int m0 = (blockIdx.x*GMT + it)*16;
    if (it + 1 < GMT) {  // prefetch next A tile (global_prefetch_b8)
      const void* nxt = A16 ? (const void*)(A16 + (size_t)(m0+16)*32)
                            : (const void*)(Zin + (size_t)(m0+16)*32);
      __builtin_prefetch(nxt, 0, 3);
    }
    v16h a = A16 ? load_a_frag(A16, m0) : load_a_bn(Zin, m0, stats, g, be);
    v8f c = {};
    c = __builtin_amdgcn_wmma_f32_16x16x32_f16(false, a, false, b, (short)0, c, false, false);
    int mb = m0 + ((lane & 16) ? 8 : 0);
    #pragma unroll
    for (int r = 0; r < 8; r++) Z[(size_t)(mb+r)*Nout + n] = c[r] + bv;
  }
}

// ---------------- batch-norm batch statistics ----------------
__global__ void k_bnstat(const float* __restrict__ Z, int ncol, int nrow, float* __restrict__ stats) {
  int c = blockIdx.x;
  float s = 0.f, s2 = 0.f;
  for (int r = threadIdx.x; r < nrow; r += blockDim.x) {
    float v = Z[(size_t)r*ncol + c]; s += v; s2 += v*v;
  }
  __shared__ float sh[256], sh2[256];
  sh[threadIdx.x] = s; sh2[threadIdx.x] = s2; __syncthreads();
  for (int o = 128; o > 0; o >>= 1) {
    if ((int)threadIdx.x < o) { sh[threadIdx.x] += sh[threadIdx.x+o]; sh2[threadIdx.x] += sh2[threadIdx.x+o]; }
    __syncthreads();
  }
  if (threadIdx.x == 0) {
    float mu = sh[0] / (float)nrow;
    float var = sh2[0] / (float)nrow - mu*mu;
    stats[c] = mu; stats[ncol + c] = rsqrtf(var + 1e-5f);
  }
}

// ---------------- fused: R = act(Z2) @ w3^T (WMMA, LDS-staged) + equivariant message ----------------
// Y tile (16 edges x 49 floats, contiguous) staged into LDS via the Tensor Data Mover
// (1-D D# descriptor, wave0 issues, s_wait_tensorcnt); cooperative-copy fallback.
__global__ void k_msg(const float* __restrict__ Z2, const float* __restrict__ stats,
                      const float* __restrict__ g, const float* __restrict__ be,
                      const _Float16* __restrict__ W3h, const float* __restrict__ b3,
                      int mo, int mi, int nj, int di, int dq, int chunkC, int cols,
                      const float* __restrict__ qjp, const float* __restrict__ Y,
                      const float* __restrict__ hsrc, const int* __restrict__ src,
                      float* __restrict__ outp, int ostride, int cstride, int ooff, int accum) {
  extern __shared__ float lds[];
  int dimi = 2*di+1, dimo = 2*dq+1;
  int bsz = dimo*dimi*nj, hsz = mi*dimi;
  float* Rl = lds;
  float* Bl = Rl + 16*cols;
  float* Hl = Bl + 16*bsz;
  float* Yl = Hl + 16*hsz;          // 16*49 floats
  int e16 = blockIdx.x * 16;

  // gather h[di][src[e]] into LDS
  for (int t = threadIdx.x; t < 16*hsz; t += blockDim.x) {
    int e = t / hsz, q = t % hsz;
    Hl[t] = hsrc[(size_t)src[e16 + e]*hsz + q];
  }

  // stage Y tile (784 contiguous floats) into LDS
#if defined(__gfx1250__) && __has_builtin(__builtin_amdgcn_tensor_load_to_lds) && __has_builtin(__builtin_amdgcn_s_wait_tensorcnt)
  if (threadIdx.x < 32) {  // wave 0 only; TDM is a wave-level op (EXEC ignored)
    unsigned ldsoff = (unsigned)((16*cols + 16*bsz + 16*hsz) * 4);  // byte offset of Yl (dynamic LDS base = 0)
    unsigned long long ga = (unsigned long long)(size_t)(const void*)(Y + (size_t)e16*49);
    u32x4 g0;
    g0[0] = 1u;                                        // count=1: valid user descriptor
    g0[1] = ldsoff;                                    // lds_addr (bytes)
    g0[2] = (unsigned)(ga & 0xFFFFFFFFull);            // global_addr[31:0]
    g0[3] = (unsigned)((ga >> 32) & 0x01FFFFFFull) | (2u << 30);  // global_addr[56:32] | type=2
    i32x8 g1;
    g1[0] = 2 << 16;                 // data_size = 4 bytes; no multicast / barrier / pad
    g1[1] = (784 & 0xFFFF) << 16;    // tensor_dim0[15:0] = 784
    g1[2] = (784 >> 16) | (1 << 16); // tensor_dim0[31:16] | tensor_dim1 = 1
    g1[3] = 784 << 16;               // tensor_dim1[31:16]=0 | tile_dim0 = 784
    g1[4] = 0;                       // tile_dim1 = 0 (1-D), tile_dim2 = 0
    g1[5] = 784;                     // tensor_dim0_stride[31:0]
    g1[6] = 0; g1[7] = 0;
    i32x4 z4 = {0, 0, 0, 0};         // groups 2/3 unused (<=2-D tensor)
#if __clang_major__ >= 23
    i32x8 z8 = {0, 0, 0, 0, 0, 0, 0, 0};
    __builtin_amdgcn_tensor_load_to_lds(g0, g1, z4, z4, z8, 0);
#else
    __builtin_amdgcn_tensor_load_to_lds(g0, g1, z4, z4, 0);
#endif
    __builtin_amdgcn_s_wait_tensorcnt(0);
  }
#else
  for (int t = threadIdx.x; t < 16*49; t += blockDim.x)
    Yl[t] = Y[(size_t)e16*49 + t];
#endif
  __syncthreads();

  // basis[e,o,i,j] = sum_m Y_J[e,m] * QJ[i,m,o]   (Y read from LDS)
  int Jlo = di > dq ? di - dq : dq - di;
  for (int t = threadIdx.x; t < 16*dimo*dimi; t += blockDim.x) {
    int e = t / (dimo*dimi); int rem = t % (dimo*dimi);
    int o = rem / dimi, i = rem % dimi;
    const float* Ye = Yl + e*49;
    int joff = 0;
    for (int j = 0; j < nj; j++) {
      int J = Jlo + j, dj = 2*J+1;
      const float* Q = qjp + joff;
      const float* Yj = Ye + J*J;
      float s = 0.f;
      for (int m = 0; m < dj; m++) s += Yj[m] * Q[(i*dj + m)*dimo + o];
      Bl[((size_t)e*dimo + o)*dimi*nj + i*nj + j] = s;
      joff += dimi*dj*dimo;
    }
  }

  v16h a = load_a_bn(Z2, e16, stats, g, be);   // BN+ReLU fused into A fragment
  int wave = threadIdx.x >> 5;
  int cgrp = mi * nj;
  for (int c0 = 0; c0 < mo; c0 += chunkC) {
    int cn = (c0 + chunkC <= mo) ? chunkC : (mo - c0);
    int coff = c0 * cgrp;
    int ncols = cn * cgrp;
    __syncthreads();  // R reuse across chunks + make Bl visible
    for (int nt = wave; nt < (ncols >> 4); nt += 4) {
      v16h b = load_a_frag(W3h, coff + nt*16);
      v8f c = {};
      c = __builtin_amdgcn_wmma_f32_16x16x32_f16(false, a, false, b, (short)0, c, false, false);
      int lane = threadIdx.x & 31;
      int ncl = nt*16 + (lane & 15);
      int mb = (lane & 16) ? 8 : 0;
      float bv = b3[coff + ncl];
      #pragma unroll
      for (int r = 0; r < 8; r++) Rl[(size_t)(mb + r)*cols + ncl] = c[r] + bv;
    }
    __syncthreads();
    // m[e,c,o] = sum_{d,i,j} R[e,c,d,j] * basis[e,o,i,j] * h[e,d,i]
    for (int t = threadIdx.x; t < 16*cn*dimo; t += blockDim.x) {
      int e = t / (cn*dimo); int rem = t % (cn*dimo);
      int cl = rem / dimo, o = rem % dimo;
      int c = c0 + cl;
      const float* Re = Rl + (size_t)e*cols + cl*cgrp;
      const float* Be = Bl + ((size_t)e*dimo + o)*dimi*nj;
      const float* He = Hl + (size_t)e*hsz;
      float acc = 0.f;
      for (int d = 0; d < mi; d++)
        for (int i = 0; i < dimi; i++) {
          float hv = He[d*dimi + i];
          for (int j = 0; j < nj; j++)
            acc += Re[d*nj + j] * Be[i*nj + j] * hv;
        }
      float* dp = outp + (size_t)(e16 + e)*ostride + c*cstride + ooff + o;
      if (accum) *dp += acc; else *dp = acc;
    }
  }
}

// ---------------- attention ----------------
__global__ void k_qlin(const float* __restrict__ qw, const float* __restrict__ h,
                       int mi, int dim, float* __restrict__ qcat, int ooff) {
  int n = blockIdx.x;
  for (int t = threadIdx.x; t < NHEADS*dim; t += blockDim.x) {
    int c = t / dim, o = t % dim;
    float s = 0.f;
    for (int k = 0; k < mi; k++) s += qw[c*mi + k] * h[((size_t)n*mi + k)*dim + o];
    qcat[((size_t)n*NHEADS + c)*16 + ooff + o] = s;
  }
}

__global__ void k_attn(const float* __restrict__ kcat, const float* __restrict__ qcat,
                       int ncat, float scale, float* __restrict__ alpha) {
  int n = blockIdx.x;
  int h = threadIdx.x >> 4, t = threadIdx.x & 15;
  int e = n*16 + t;
  float s = 0.f;
  for (int k = 0; k < ncat; k++)
    s += kcat[((size_t)e*NHEADS + h)*16 + k] * qcat[((size_t)n*NHEADS + h)*16 + k];
  s *= scale;
  __shared__ float sh[64];
  sh[threadIdx.x] = s; __syncthreads();
  float mx = -3.4e38f;
  for (int i = 0; i < 16; i++) mx = fmaxf(mx, sh[h*16 + i]);
  float ex = expf(s - mx);
  __syncthreads();
  sh[threadIdx.x] = ex; __syncthreads();
  float den = 0.f;
  for (int i = 0; i < 16; i++) den += sh[h*16 + i];
  alpha[(size_t)e*NHEADS + h] = ex / den;
}

// ---------------- node update: aggregate + proj + equivariant norm-nonlinearity ----------------
__global__ void k_nodeupd(const float* __restrict__ vout, const float* __restrict__ alpha,
                          const float* __restrict__ hprev, int mip,
                          const float* __restrict__ proj, int cat_in,
                          const float* __restrict__ lng, const float* __restrict__ lnb,
                          float* __restrict__ hnew, int dim) {
  int n = blockIdx.x;
  __shared__ float zc[(MO_MID + NODE_FEAT) * 7];
  __shared__ float nr[NCH];
  __shared__ float mb[2];
  for (int t = threadIdx.x; t < MO_MID*dim; t += blockDim.x) {
    int c = t / dim, m = t % dim;
    float s = 0.f;
    for (int e = 0; e < 16; e++) {
      int eg = n*16 + e;
      s += alpha[(size_t)eg*NHEADS + c] * vout[((size_t)eg*MO_MID + c)*dim + m];
    }
    zc[c*dim + m] = s;
  }
  for (int t = threadIdx.x; t < mip*dim; t += blockDim.x) {
    int ch = t / dim, m = t % dim;
    zc[(MO_MID + ch)*dim + m] = hprev[((size_t)n*mip + ch)*dim + m];
  }
  __syncthreads();
  float row[7];
  int o = threadIdx.x;
  if (o < NCH) {
    float ss = 0.f;
    for (int m = 0; m < dim; m++) {
      float s = 0.f;
      for (int c = 0; c < cat_in; c++) s += proj[o*cat_in + c] * zc[c*dim + m];
      row[m] = s; ss += s*s;
    }
    nr[o] = sqrtf(ss + 1e-12f);
  }
  __syncthreads();
  if (threadIdx.x == 0) {
    float mu = 0.f; for (int i = 0; i < NCH; i++) mu += nr[i]; mu *= (1.f/NCH);
    float var = 0.f; for (int i = 0; i < NCH; i++) { float d = nr[i]-mu; var += d*d; } var *= (1.f/NCH);
    mb[0] = mu; mb[1] = rsqrtf(var + 1e-5f);
  }
  __syncthreads();
  if (o < NCH) {
    float tg = fmaxf((nr[o]-mb[0])*mb[1]*lng[o] + lnb[o], 0.f);
    float inv = 1.f / nr[o];
    for (int m = 0; m < dim; m++) hnew[((size_t)n*NCH + o)*dim + m] = tg * row[m] * inv;
  }
}

// ---------------- final stages ----------------
__global__ void k_finalnode(const float* __restrict__ acc, const float* __restrict__ selfw,
                            const float* __restrict__ h0, float* __restrict__ h0n) {
  int n = blockIdx.x, c = threadIdx.x;
  float s = 0.f;
  for (int e = 0; e < 16; e++) s += acc[((size_t)(n*16 + e))*OUTD + c];
  s *= (1.f/16.f);
  for (int k = 0; k < NCH; k++) s += selfw[c*NCH + k] * h0[(size_t)n*NCH + k];
  h0n[(size_t)n*OUTD + c] = s;
}

__global__ void k_meanpool(const float* __restrict__ h0n, float* __restrict__ pooled) {
  int c = blockIdx.x;
  float s = 0.f;
  for (int n = threadIdx.x; n < NNODES; n += blockDim.x) s += h0n[(size_t)n*OUTD + c];
  __shared__ float sh[256];
  sh[threadIdx.x] = s; __syncthreads();
  for (int o = 128; o > 0; o >>= 1) { if ((int)threadIdx.x < o) sh[threadIdx.x] += sh[threadIdx.x+o]; __syncthreads(); }
  if (threadIdx.x == 0) pooled[c] = sh[0] / (float)NNODES;
}

__global__ void k_fc(const float* __restrict__ pooled,
                     const float* __restrict__ w1, const float* __restrict__ b1,
                     const float* __restrict__ w2, const float* __restrict__ b2,
                     float* __restrict__ out) {
  __shared__ float z[OUTD];
  int t = threadIdx.x;
  float s = b1[t];
  for (int k = 0; k < OUTD; k++) s += w1[t*OUTD + k] * pooled[k];
  z[t] = fmaxf(s, 0.f);
  __syncthreads();
  float s2 = b2[t];
  for (int k = 0; k < OUTD; k++) s2 += w2[t*OUTD + k] * z[k];
  out[t] = s2;
}

// ---------------- parameter index map (tree_flatten, sorted dict keys) ----------------
enum { RP_B1=0, RP_B2, RP_B3, RP_BE1, RP_BE2, RP_G1, RP_G2, RP_W1, RP_W2, RP_W3 };
static const int IDX_FC1B=5, IDX_FC1W=6, IDX_FC2B=7, IDX_FC2W=8;
static const int L0_K=9,  L0_LNB=19,  L0_LNG=23,  L0_PROJ=27,  L0_Q=31,  L0_V=32;
static const int L1_K=72, L1_LNB=232, L1_LNG=236, L1_PROJ=240, L1_Q=244, L1_V=248;
static const int OUT_R=408, OUT_SELF=448;

extern "C" void kernel_launch(void* const* d_in, const int* in_sizes, int n_in,
                              void* d_out, int out_size, void* d_ws, size_t ws_size,
                              hipStream_t stream) {
  (void)in_sizes; (void)out_size; (void)ws_size;
  auto P = [&](int idx) -> const float* {
    return (const float*)d_in[(idx >= 0 && idx < n_in) ? idx : 0];
  };
  const float* x     = (const float*)d_in[0];
  const float* fin   = (const float*)d_in[1];
  const float* wfeat = (const float*)d_in[2];
  const int*   src   = (const int*)d_in[3];
  const int*   dstp  = (const int*)d_in[4];

  char* cur = (char*)d_ws;
  auto alloc = [&](size_t bytes) -> void* {
    void* p = (void*)cur;
    cur += (bytes + 255) & ~(size_t)255;
    return p;
  };
  float*    qj    = (float*)alloc(7056*sizeof(float));
  float*    Y     = (float*)alloc((size_t)NEDGES*49*sizeof(float));
  _Float16* featP = (_Float16*)alloc((size_t)NEDGES*32*sizeof(_Float16));
  float*    Z1    = (float*)alloc((size_t)NEDGES*32*sizeof(float));
  float*    Z2    = (float*)alloc((size_t)NEDGES*32*sizeof(float));
  _Float16* w1h   = (_Float16*)alloc((size_t)32*32*sizeof(_Float16));
  _Float16* w2h   = (_Float16*)alloc((size_t)32*32*sizeof(_Float16));
  _Float16* w3h   = (_Float16*)alloc((size_t)1024*32*sizeof(_Float16));
  float*    st1   = (float*)alloc(64*sizeof(float));
  float*    st2   = (float*)alloc(64*sizeof(float));
  float* vout[4];
  for (int d = 0; d < 4; d++) vout[d] = (float*)alloc((size_t)NEDGES*MO_MID*(2*d+1)*sizeof(float));
  float* kcat   = (float*)alloc((size_t)NEDGES*64*sizeof(float));
  float* qcat   = (float*)alloc((size_t)NNODES*64*sizeof(float));
  float* alphab = (float*)alloc((size_t)NEDGES*NHEADS*sizeof(float));
  float* hA[4]; float* hB[4];
  for (int d = 0; d < 4; d++) hA[d] = (float*)alloc((size_t)NNODES*NCH*7*sizeof(float));
  for (int d = 0; d < 4; d++) hB[d] = (float*)alloc((size_t)NNODES*NCH*7*sizeof(float));
  float* accb   = (float*)alloc((size_t)NEDGES*OUTD*sizeof(float));
  float* h0n    = (float*)alloc((size_t)NNODES*OUTD*sizeof(float));
  float* pooled = (float*)alloc(64*sizeof(float));

  k_qj<<<1, 64, 0, stream>>>(qj);
  k_edge<<<NEDGES/256, 256, 0, stream>>>(x, wfeat, src, dstp, Y, featP);

  // radial MLP (5->32->32): f16 weights, WMMA GEMMs, BN fused into consumer A-loads
  auto run_radial = [&](int base, int Cout) {
    k_cvtw<<<4, 256, 0, stream>>>(P(base+RP_W1), 5, 5, 32, w1h);
    k_cvtw<<<4, 256, 0, stream>>>(P(base+RP_W2), 32, 32, 32, w2h);
    k_cvtw<<<(Cout*32 + 255)/256, 256, 0, stream>>>(P(base+RP_W3), 32, 32, Cout, w3h);
    k_gemm16<<<dim3(NEDGES/(16*GMT), 2), 32, 0, stream>>>(
        featP, nullptr, st1, P(base+RP_G1), P(base+RP_BE1), w1h, P(base+RP_B1), Z1, 32);
    k_bnstat<<<32, 256, 0, stream>>>(Z1, 32, NEDGES, st1);
    k_gemm16<<<dim3(NEDGES/(16*GMT), 2), 32, 0, stream>>>(
        nullptr, Z1, st1, P(base+RP_G1), P(base+RP_BE1), w2h, P(base+RP_B2), Z2, 32);
    k_bnstat<<<32, 256, 0, stream>>>(Z2, 32, NEDGES, st2);
  };

  auto run_msg = [&](int base, int di, int dq, int mi, int mo, const float* hsrc,
                     float* outp, int ostride, int cstride, int ooff, int accum) {
    int nj = 2*(di < dq ? di : dq) + 1;
    int Cout = nj * mi * mo;
    run_radial(base, Cout);
    int dimi = 2*di+1, dimo = 2*dq+1;
    int cgrp = mi * nj;
    int cc = 512 / cgrp; if (cc < 1) cc = 1; if (cc > mo) cc = mo;
    int cols = cc * cgrp;
    int bsz = dimo*dimi*nj, hsz = mi*dimi;
    size_t ldsb = (size_t)(16*cols + 16*bsz + 16*hsz + 16*49) * sizeof(float);
    k_msg<<<NEDGES/16, 128, ldsb, stream>>>(Z2, st2, P(base+RP_G2), P(base+RP_BE2),
        w3h, P(base+RP_B3),
        mo, mi, nj, di, dq, cc, cols,
        qj + qj_pair_off(di, dq), Y, hsrc, src,
        outp, ostride, cstride, ooff, accum);
  };

  const float* hcur[4] = { fin, nullptr, nullptr, nullptr };
  int micur[4] = { NODE_FEAT, 0, 0, 0 };
  const int catoff[4] = { 0, 1, 4, 9 };

  for (int L = 0; L < NLAYERS; L++) {
    int KB = L ? L1_K : L0_K,   VB = L ? L1_V : L0_V,     QB = L ? L1_Q : L0_Q;
    int PB = L ? L1_PROJ : L0_PROJ, GB = L ? L1_LNG : L0_LNG, BB = L ? L1_LNB : L0_LNB;
    // value messages: f_out = all degrees
    for (int dq = 0; dq < 4; dq++) {
      int accf = 0;
      for (int di = 0; di < 4; di++) if (hcur[di]) {
        int pi = L ? (di*4 + dq) : dq;
        run_msg(VB + pi*10, di, dq, micur[di], MO_MID, hcur[di],
                vout[dq], MO_MID*(2*dq+1), (2*dq+1), 0, accf);
        accf = 1;
      }
    }
    // key messages: f_out = f_in degrees; write directly into concat layout
    for (int dq = 0; dq < 4; dq++) if (hcur[dq]) {
      int accf = 0;
      for (int di = 0; di < 4; di++) if (hcur[di]) {
        int pi = L ? (di*4 + dq) : 0;
        run_msg(KB + pi*10, di, dq, micur[di], MO_MID, hcur[di],
                kcat, 64, 16, catoff[dq], accf);
        accf = 1;
      }
    }
    // queries + attention
    int ncat = 0, dk = 0;
    for (int d = 0; d < 4; d++) if (hcur[d]) {
      k_qlin<<<NNODES, 32, 0, stream>>>(P(QB + (L ? d : 0)), hcur[d], micur[d], 2*d+1, qcat, catoff[d]);
      ncat += 2*d+1; dk += MO_MID*(2*d+1);
    }
    k_attn<<<NNODES, 64, 0, stream>>>(kcat, qcat, ncat, 1.f/sqrtf((float)dk), alphab);
    // node update per degree
    float** hnxt = (L == 0) ? hA : hB;
    for (int d = 0; d < 4; d++) {
      int mip = hcur[d] ? micur[d] : 0;
      k_nodeupd<<<NNODES, 64, 0, stream>>>(vout[d], alphab, hcur[d], mip,
          P(PB + d), MO_MID + mip, P(GB + d), P(BB + d), hnxt[d], 2*d+1);
    }
    for (int d = 0; d < 4; d++) { hcur[d] = hnxt[d]; micur[d] = NCH; }
  }

  // output block: degree-0 messages with OUT_DIM channels
  for (int di = 0; di < 4; di++)
    run_msg(OUT_R + di*10, di, 0, NCH, OUTD, hcur[di], accb, OUTD, 1, 0, di > 0 ? 1 : 0);
  k_finalnode<<<NNODES, 64, 0, stream>>>(accb, P(OUT_SELF), hcur[0], h0n);
  k_meanpool<<<64, 256, 0, stream>>>(h0n, pooled);
  k_fc<<<1, 64, 0, stream>>>(pooled, P(IDX_FC1W), P(IDX_FC1B), P(IDX_FC2W), P(IDX_FC2B), (float*)d_out);
}
